// TrulyNeuralSyscallHandlersV3_32512902431250
// MI455X (gfx1250) — matrix operand order repaired and checked by reading, not compiled, not run on hardware
//
#include <hip/hip_runtime.h>
#include <hip/hip_bf16.h>
#include <math.h>

typedef __attribute__((ext_vector_type(16))) _Float16 v16h;
typedef __attribute__((ext_vector_type(8)))  float    v8f;

#define NSYS   512
#define IN_DIM 400
#define K_PAD  416          // 13 * 32
#define KC1    13
#define HID    128
#define KC2    4            // 128 / 32
#define OUTD   64           // only out[...,:64] is used downstream
#define RPW    128          // rows per workgroup (8 waves x 16 rows)

// fragment-major packed weights: [e][kc][nt][lane][16 halves]
#define W1F_HALVES (8 * KC1 * 8 * 32 * 16)   // 425,984 halves
#define W2F_HALVES (8 * KC2 * 4 * 32 * 16)   //  65,536 halves

// ---------------------------------------------------------------------------
// W1 (8 x 400 x 128 f32) -> f16 B-fragments, K padded to 416 with zeros.
// B 32x16 f16 layout: half h of lane L holds K = kc*32 + (L>=16 ? 8 : 0) + (h<8 ? h : 16+h-8),
// N = nt*16 + (L & 15).
// ---------------------------------------------------------------------------
__global__ void prep_w1_frag(const float* __restrict__ W1, _Float16* __restrict__ w1f) {
    int idx = blockIdx.x * 256 + threadIdx.x;
    if (idx >= W1F_HALVES) return;
    int h    = idx & 15;
    int t    = idx >> 4;
    int lane = t & 31;  t >>= 5;
    int nt   = t & 7;   t >>= 3;
    int kc   = t % KC1;
    int e    = t / KC1;
    int k = kc * 32 + ((lane >= 16) ? 8 : 0) + ((h < 8) ? h : (16 + h - 8));
    int n = nt * 16 + (lane & 15);
    float v = (k < IN_DIM) ? W1[((size_t)e * IN_DIM + k) * HID + n] : 0.0f;
    w1f[idx] = (_Float16)v;
}

// W2 (8 x 128 x 65 f32) -> f16 B-fragments over the first 64 output columns.
__global__ void prep_w2_frag(const float* __restrict__ W2, _Float16* __restrict__ w2f) {
    int idx = blockIdx.x * 256 + threadIdx.x;
    if (idx >= W2F_HALVES) return;
    int h    = idx & 15;
    int t    = idx >> 4;
    int lane = t & 31;  t >>= 5;
    int nt   = t & 3;   t >>= 2;
    int kc   = t & 3;
    int e    = t >> 2;
    int k = kc * 32 + ((lane >= 16) ? 8 : 0) + ((h < 8) ? h : (16 + h - 8));   // < 128 always
    int n = nt * 16 + (lane & 15);                   // < 64 always
    w2f[idx] = (_Float16)W2[((size_t)e * HID + k) * 65 + n];
}

// ---------------------------------------------------------------------------
// Pack activations x = concat(sys_bits, arg_bits) into f16 A-fragment layout:
// xf[((rt*13 + kc)*32 + lane)*16 + h].  A 16x32 f16 layout: lane L holds row
// m = L&15; half h (VGPR g=h/2, pos p=h&1) holds
// k = kc*32 + (L>=16 ? 8 : 0) + (g<4 ? 2g+p : 16+2(g-4)+p).
// ---------------------------------------------------------------------------
__global__ void prep_x_frag(const float* __restrict__ sys_bits,
                            const float* __restrict__ arg_bits,
                            _Float16* __restrict__ xf, int ntiles) {
    int idx = blockIdx.x * 256 + threadIdx.x;
    int h    = idx & 15;
    int t    = idx >> 4;
    int lane = t & 31;  t >>= 5;
    int kc   = t % KC1;
    int rt   = t / KC1;
    if (rt >= ntiles) return;
    int m    = lane & 15;
    int koff = (lane >= 16) ? 8 : 0;
    int g = h >> 1, p = h & 1;
    int k = kc * 32 + koff + ((g < 4) ? (2 * g + p) : (16 + 2 * (g - 4) + p));
    int row = rt * 16 + m;
    float v = 0.0f;
    if (k < 16)          v = sys_bits[(size_t)row * 16 + k];
    else if (k < IN_DIM) v = arg_bits[(size_t)row * 384 + (k - 16)];
    xf[idx] = (_Float16)v;
}

// ---------------------------------------------------------------------------
// Main fused kernel: 256 threads = 8 waves; wave w owns one 16-row tile and
// iterates all 8 subsystems, so weight fragments staged in LDS feed all waves.
// ---------------------------------------------------------------------------
__global__ __launch_bounds__(256)
void handler_main(const float* __restrict__ sys_bits,   // B x 16 (routing only)
                  const float* __restrict__ rtable,     // 512 x 8
                  const float* __restrict__ temp_p,     // 1
                  const float* __restrict__ b1,         // 8 x 128
                  const float* __restrict__ b2,         // 8 x 65
                  const _Float16* __restrict__ w1f,
                  const _Float16* __restrict__ w2f,
                  const _Float16* __restrict__ xf,
                  float* __restrict__ out_w,            // B x 64
                  float* __restrict__ out_s,            // B
                  float* __restrict__ out_p)            // B x 8
{
    __shared__ v16h     bufv[2][256];        // 16KB double-buffered B staging
    __shared__ _Float16 hsc[8][16 * HID];    // 32KB per-wave hidden scratch
    __shared__ float    probs[RPW][8];       // 4KB route probs per row

    const int tid   = threadIdx.x;
    const int lane  = tid & 31;
    const int wv    = tid >> 5;
    const int gtile = blockIdx.x * 8 + wv;   // global 16-row tile of this wave
    const int row0  = blockIdx.x * RPW;

    // ---- routing softmax: thread t handles local row t (t < 128) ----
    if (tid < RPW) {
        int b = row0 + tid;
        int s = 0;
        #pragma unroll
        for (int i = 0; i < 16; ++i)
            s |= (sys_bits[(size_t)b * 16 + i] > 0.5f) ? (1 << i) : 0;
        if (s > NSYS - 1) s = NSYS - 1;
        float tinv = 1.0f / fmaxf(fabsf(temp_p[0]), 0.1f);
        float l[8], mx = -3.0e38f;
        #pragma unroll
        for (int e = 0; e < 8; ++e) { l[e] = rtable[s * 8 + e] * tinv; mx = fmaxf(mx, l[e]); }
        float sum = 0.0f;
        #pragma unroll
        for (int e = 0; e < 8; ++e) { l[e] = expf(l[e] - mx); sum += l[e]; }
        float inv = 1.0f / sum, best = 0.0f;
        #pragma unroll
        for (int e = 0; e < 8; ++e) {
            float p = l[e] * inv;
            probs[tid][e] = p;
            out_p[(size_t)b * 8 + e] = p;
            best = fmaxf(best, p);
        }
        out_s[b] = best;
    }

    // ---- load this wave's 13 A-fragments (prepacked, contiguous 32B/lane) ----
    const v16h* xfv  = (const v16h*)xf;
    const v16h* w1fv = (const v16h*)w1f;
    const v16h* w2fv = (const v16h*)w2f;
    v16h a1[KC1];
    #pragma unroll
    for (int kc = 0; kc < KC1; ++kc)
        a1[kc] = xfv[((size_t)gtile * KC1 + kc) * 32 + lane];

    const int mrow = (lane >= 16) ? 8 : 0;
    const int ncol = lane & 15;
    const int m    = lane & 15;
    const int koff = (lane >= 16) ? 8 : 0;

    v8f wsum[4];
    #pragma unroll
    for (int nt = 0; nt < 4; ++nt)
        #pragma unroll
        for (int i = 0; i < 8; ++i) wsum[nt][i] = 0.0f;

    // preload W1 fragments for (e=0, kc=0)
    bufv[0][tid] = w1fv[(size_t)tid];
    __syncthreads();

    for (int e = 0; e < 8; ++e) {
        // ---- layer 1: acc = x @ W1[e], double-buffered LDS weight broadcast ----
        v8f acc[8];
        #pragma unroll
        for (int nt = 0; nt < 8; ++nt)
            #pragma unroll
            for (int i = 0; i < 8; ++i) acc[nt][i] = 0.0f;

        for (int kc = 0; kc < KC1; ++kc) {
            v16h nxt;
            bool more = (kc + 1 < KC1);
            if (more) nxt = w1fv[((size_t)(e * KC1 + kc + 1) * 8) * 32 + tid];
            int cur = kc & 1;
            #pragma unroll
            for (int nt = 0; nt < 8; ++nt) {
                v16h bb = bufv[cur][nt * 32 + lane];
                acc[nt] = __builtin_amdgcn_wmma_f32_16x16x32_f16(
                    false, a1[kc], false, bb, (short)0, acc[nt], false, false);
            }
            if (more) bufv[cur ^ 1][tid] = nxt;
            __syncthreads();
        }

        // ---- bias + exact gelu, write h to this wave's private LDS scratch ----
        #pragma unroll
        for (int nt = 0; nt < 8; ++nt) {
            int n = nt * 16 + ncol;
            float bias = b1[e * HID + n];
            #pragma unroll
            for (int r = 0; r < 8; ++r) {
                float x = acc[nt][r] + bias;
                float g = 0.5f * x * (1.0f + erff(x * 0.70710678118654752f));
                hsc[wv][(mrow + r) * HID + n] = (_Float16)g;
            }
        }

        // ---- re-fragment h (wave-private LDS, DS ops are in-order per wave) ----
        v16h a2[KC2];
        #pragma unroll
        for (int kc = 0; kc < KC2; ++kc)
            #pragma unroll
            for (int h = 0; h < 16; ++h) {
                int g = h >> 1, p = h & 1;
                int k = kc * 32 + koff + ((g < 4) ? (2 * g + p) : (16 + 2 * (g - 4) + p));
                a2[kc][h] = hsc[wv][m * HID + k];
            }

        // ---- layer 2: out = h @ W2[e] (64 cols), same staging scheme ----
        v8f acc2[4];
        #pragma unroll
        for (int nt = 0; nt < 4; ++nt)
            #pragma unroll
            for (int i = 0; i < 8; ++i) acc2[nt][i] = 0.0f;

        __syncthreads();
        if (tid < 128) bufv[0][tid] = w2fv[((size_t)(e * KC2) * 4) * 32 + tid];
        __syncthreads();
        for (int kc = 0; kc < KC2; ++kc) {
            v16h nxt;
            bool more = (kc + 1 < KC2);
            if (more && tid < 128) nxt = w2fv[((size_t)(e * KC2 + kc + 1) * 4) * 32 + tid];
            int cur = kc & 1;
            #pragma unroll
            for (int nt = 0; nt < 4; ++nt) {
                v16h bb = bufv[cur][nt * 32 + lane];
                acc2[nt] = __builtin_amdgcn_wmma_f32_16x16x32_f16(
                    false, a2[kc], false, bb, (short)0, acc2[nt], false, false);
            }
            if (more && tid < 128) bufv[cur ^ 1][tid] = nxt;
            __syncthreads();
        }

        // ---- weighted accumulation over subsystems, entirely in registers ----
        #pragma unroll
        for (int nt = 0; nt < 4; ++nt) {
            int n = nt * 16 + ncol;
            float bias = b2[e * 65 + n];
            #pragma unroll
            for (int r = 0; r < 8; ++r) {
                float p = probs[wv * 16 + mrow + r][e];
                wsum[nt][r] += p * (acc2[nt][r] + bias);
            }
        }

        // preload W1 fragments for next e, kc=0 (buffer 0 free after barrier)
        if (e + 1 < 8) {
            bufv[0][tid] = w1fv[((size_t)((e + 1) * KC1) * 8) * 32 + tid];
            __syncthreads();
        }
    }

    // ---- write weighted result (C-layout lanes -> 64B segments per row) ----
    #pragma unroll
    for (int nt = 0; nt < 4; ++nt) {
        int n = nt * 16 + ncol;
        #pragma unroll
        for (int r = 0; r < 8; ++r) {
            int row = gtile * 16 + mrow + r;
            out_w[(size_t)row * OUTD + n] = wsum[nt][r];
        }
    }
}

extern "C" void kernel_launch(void* const* d_in, const int* in_sizes, int n_in,
                              void* d_out, int out_size, void* d_ws, size_t ws_size,
                              hipStream_t stream) {
    const float* sys_bits = (const float*)d_in[0];   // (B,16)
    const float* arg_bits = (const float*)d_in[1];   // (B,6,64)
    const float* rtable   = (const float*)d_in[2];   // (512,8)
    const float* temp     = (const float*)d_in[3];   // scalar
    const float* W1       = (const float*)d_in[4];   // (8,400,128)
    const float* b1       = (const float*)d_in[5];   // (8,128)
    const float* W2       = (const float*)d_in[6];   // (8,128,65)
    const float* b2       = (const float*)d_in[7];   // (8,65)

    const int B      = in_sizes[0] / 16;
    const int ntiles = B / 16;

    _Float16* w1f = (_Float16*)d_ws;                     //   851,968 B
    _Float16* w2f = w1f + W1F_HALVES;                    //   131,072 B
    _Float16* xfr = w2f + W2F_HALVES;                    // ~109 MB: (B/16)*13*512 halves

    float* out   = (float*)d_out;
    float* out_w = out;                                  // B*64
    float* out_s = out + (size_t)B * 64;                 // B
    float* out_p = out + (size_t)B * 65;                 // B*8

    hipLaunchKernelGGL(prep_w1_frag, dim3((W1F_HALVES + 255) / 256), dim3(256), 0, stream, W1, w1f);
    hipLaunchKernelGGL(prep_w2_frag, dim3((W2F_HALVES + 255) / 256), dim3(256), 0, stream, W2, w2f);
    {
        long long tot = (long long)ntiles * KC1 * 32 * 16;
        hipLaunchKernelGGL(prep_x_frag, dim3((unsigned)((tot + 255) / 256)), dim3(256), 0, stream,
                           sys_bits, arg_bits, xfr, ntiles);
    }
    hipLaunchKernelGGL(handler_main, dim3(B / RPW), dim3(256), 0, stream,
                       sys_bits, rtable, temp, b1, b2, w1f, w2f, xfr,
                       out_w, out_s, out_p);
}
